// GPT2Attention_5884105195726
// MI455X (gfx1250) — matrix operand order repaired and checked by reading, not compiled, not run on hardware
//
#include <hip/hip_runtime.h>

// ---------------------------------------------------------------------------
// GPT-2 attention for MI455X (gfx1250, wave32, WMMA f32_16x16x32_f16)
//   B=2, S=2048, D=1024, H=16, DK=64. No 1/sqrt(dk) scaling (matches ref).
// Pipeline:
//   1) f32->f16 convert (activations) + convert-transpose (weights)
//   2) QKV GEMM (WMMA) -> scatter Q,K [B,H,S,64], V^T [B,H,64,S]
//   3) causal flash attention (WMMA)
//   4) out-proj GEMM (WMMA) -> f32
// Staging uses CDNA5 async global->LDS (GLOBAL_LOAD_ASYNC_TO_LDS_B128,
// ASYNCcnt) with LDS double buffering so prefetch overlaps WMMA compute.
// ---------------------------------------------------------------------------

typedef __attribute__((ext_vector_type(16))) _Float16 v16h;
typedef __attribute__((ext_vector_type(8)))  float    v8f;

union F16Frag { v16h v; uint4 q[2]; };

#define WMMA16x16x32(Afr, Bfr, C) \
    __builtin_amdgcn_wmma_f32_16x16x32_f16(false, (Afr), false, (Bfr), (short)0, (C), false, false)

// 64 bytes per lane: LDS dest (32-bit addr) and global src advance together
// (ISA: INST_OFFSET is added to BOTH the LDS and global address).
__device__ __forceinline__ void async_copy64B(unsigned lds, const void* gp) {
    asm volatile(
        "global_load_async_to_lds_b128 %0, %1, off\n\t"
        "global_load_async_to_lds_b128 %0, %1, off offset:16\n\t"
        "global_load_async_to_lds_b128 %0, %1, off offset:32\n\t"
        "global_load_async_to_lds_b128 %0, %1, off offset:48"
        :: "v"(lds), "v"(gp) : "memory");
}
#define ASYNC_WAIT0() asm volatile("s_wait_asynccnt 0x0" ::: "memory")
#define LDS_OFF(p) ((unsigned)(size_t)(p))

#define GB 2
#define GS 2048
#define GD 1024
#define GH 16
#define GDK 64
#define NEG_BIG (-3.0e38f)

// ---------------------------------------------------------------------------
// f32 -> f16 elementwise convert
// ---------------------------------------------------------------------------
__global__ __launch_bounds__(256) void cvt_f32_to_f16(const float* __restrict__ in,
                                                      _Float16* __restrict__ out, int n) {
    int i = blockIdx.x * 256 + threadIdx.x;
    if (i < n) out[i] = (_Float16)in[i];
}

// ---------------------------------------------------------------------------
// f32 [K,N] -> f16 [N,K] convert + transpose (tiled, coalesced both sides)
// grid: (N/32, K/32), block: 256 = 32x8
// ---------------------------------------------------------------------------
__global__ __launch_bounds__(256) void cvt_f32_to_f16_T(const float* __restrict__ in,
                                                        _Float16* __restrict__ out,
                                                        int K, int N) {
    __shared__ _Float16 tile[32][33];
    const int tx = threadIdx.x & 31;
    const int ty = threadIdx.x >> 5;
    const int nb = blockIdx.x * 32;
    const int kb = blockIdx.y * 32;
#pragma unroll
    for (int i = 0; i < 32; i += 8)
        tile[tx][ty + i] = (_Float16)in[(size_t)(kb + ty + i) * N + nb + tx];
    __syncthreads();
#pragma unroll
    for (int i = 0; i < 32; i += 8)
        out[(size_t)(nb + ty + i) * K + kb + tx] = tile[ty + i][tx];
}

// ---------------------------------------------------------------------------
// GEMM: C[M,N] = A[M,K](f16) * Bt[N,K](f16, pre-transposed) + bias
// mode 0: f32 to outF;  mode 1: scatter f16 Q,K [B,H,S,64] / V^T [B,H,64,S]
// Block tile 128x128, 256 thr = 8 waves (2x4), wave tile 64x32, K-step 64,
// double-buffered async global->LDS staging.
// ---------------------------------------------------------------------------
__global__ __launch_bounds__(256) void gemm_f16_wmma(
    const _Float16* __restrict__ A,   // [M,K]
    const _Float16* __restrict__ Bt,  // [N,K]
    const float*    __restrict__ bias,
    float*          __restrict__ outF,
    _Float16* __restrict__ Qo, _Float16* __restrict__ Ko, _Float16* __restrict__ Vo,
    int M, int N, int K, int mode) {
    __shared__ _Float16 As[2][128][64];  // 32 KB
    __shared__ _Float16 Bs[2][128][64];  // 32 KB

    const int tid  = threadIdx.x;
    const int lane = tid & 31;
    const int wave = tid >> 5;
    const int hi   = lane >> 4;
    const int l15  = lane & 15;
    const int wm   = (wave >> 2) * 64;
    const int wn   = (wave & 3) * 32;
    const int m0   = blockIdx.y * 128;
    const int n0   = blockIdx.x * 128;

    v8f acc[4][2];
#pragma unroll
    for (int mi = 0; mi < 4; ++mi)
#pragma unroll
        for (int ni = 0; ni < 2; ++ni) acc[mi][ni] = (v8f){0.f,0.f,0.f,0.f,0.f,0.f,0.f,0.f};

    const int srow = tid >> 1;          // stager: 128 rows x (2 x 32) cols
    const int scol = (tid & 1) * 32;
    const _Float16* gA = A  + (size_t)(m0 + srow) * K + scol;
    const _Float16* gB = Bt + (size_t)(n0 + srow) * K + scol;

    // prologue: stage k-tile 0 into buffer 0
    async_copy64B(LDS_OFF(&As[0][srow][scol]), gA);
    async_copy64B(LDS_OFF(&Bs[0][srow][scol]), gB);
    ASYNC_WAIT0();
    __syncthreads();

    int buf = 0;
    for (int k0 = 0; k0 < K; k0 += 64) {
        if (k0 + 64 < K) {  // prefetch next tile into other buffer (overlaps WMMA)
            async_copy64B(LDS_OFF(&As[buf ^ 1][srow][scol]), gA + k0 + 64);
            async_copy64B(LDS_OFF(&Bs[buf ^ 1][srow][scol]), gB + k0 + 64);
        }
#pragma unroll
        for (int kk = 0; kk < 2; ++kk) {
            F16Frag af[4], bf[2];
#pragma unroll
            for (int mi = 0; mi < 4; ++mi) {  // A: e<8 -> k=hi*8+e ; e>=8 -> 16+hi*8+(e-8)
                const int r = wm + mi * 16 + l15;
                af[mi].q[0] = *(const uint4*)&As[buf][r][kk * 32 + hi * 8];
                af[mi].q[1] = *(const uint4*)&As[buf][r][kk * 32 + 16 + hi * 8];
            }
#pragma unroll
            for (int ni = 0; ni < 2; ++ni) {  // B: e<8 -> k=hi*8+e ; e>=8 -> 16+hi*8+(e-8), col = lane&15
                const int n = wn + ni * 16 + l15;
                bf[ni].q[0] = *(const uint4*)&Bs[buf][n][kk * 32 + hi * 8];
                bf[ni].q[1] = *(const uint4*)&Bs[buf][n][kk * 32 + 16 + hi * 8];
            }
#pragma unroll
            for (int mi = 0; mi < 4; ++mi)
#pragma unroll
                for (int ni = 0; ni < 2; ++ni)
                    acc[mi][ni] = WMMA16x16x32(af[mi].v, bf[ni].v, acc[mi][ni]);
        }
        ASYNC_WAIT0();
        __syncthreads();
        buf ^= 1;
    }

    // epilogue (C layout: lane col = lane&15, row = hi*8 + vgpr r)
#pragma unroll
    for (int mi = 0; mi < 4; ++mi) {
#pragma unroll
        for (int ni = 0; ni < 2; ++ni) {
            const int n = n0 + wn + ni * 16 + l15;
            const float bv = bias[n];
#pragma unroll
            for (int r = 0; r < 8; ++r) {
                const int m = m0 + wm + mi * 16 + hi * 8 + r;
                const float v = acc[mi][ni][r] + bv;
                if (mode == 0) {
                    outF[(size_t)m * N + n] = v;
                } else {
                    const int part = n >> 10, rem = n & 1023;
                    const int h = rem >> 6, dk = rem & 63;
                    const int b = m >> 11, s = m & 2047;
                    if (part == 2) {  // V stored transposed: [bh][dk][s]
                        Vo[(((size_t)(b * GH + h) * GDK) + dk) * GS + s] = (_Float16)v;
                    } else {
                        const size_t idx = (((size_t)(b * GH + h) * GS) + s) * GDK + dk;
                        ((part == 0) ? Qo : Ko)[idx] = (_Float16)v;
                    }
                }
            }
        }
    }
}

// ---------------------------------------------------------------------------
// Causal flash attention (no scale). Q,K: [B*H, S, 64] f16; V: [B*H, 64, S].
// Block = 128 thr (4 waves), 64 query rows/block (16/wave). KV chunks of 64,
// double-buffered async global->LDS staging.
// ---------------------------------------------------------------------------
__global__ __launch_bounds__(128) void flash_attn_wmma(
    const _Float16* __restrict__ Q, const _Float16* __restrict__ K,
    const _Float16* __restrict__ V, _Float16* __restrict__ Aout) {
    const int bh  = blockIdx.x;
    const int qt  = blockIdx.y;
    const int tid = threadIdx.x;
    const int lane = tid & 31, wave = tid >> 5;
    const int hi = lane >> 4, l15 = lane & 15;

    const _Float16* Qp = Q + (size_t)bh * GS * GDK;
    const _Float16* Kp = K + (size_t)bh * GS * GDK;
    const _Float16* Vp = V + (size_t)bh * GDK * GS;  // [dk][s]

    __shared__ _Float16 Ks[2][64][64];   // [kv][dk]  16 KB
    __shared__ _Float16 Vs[2][64][64];   // [dk][kv]  16 KB
    __shared__ _Float16 Ps[4][16][64];   // per-wave   8 KB

    const int q0 = qt * 64 + wave * 16;

    // Q fragments over head dim (two K-steps of 32)
    F16Frag qf[2];
    {
        const uint4* qp = (const uint4*)(Qp + (size_t)(q0 + l15) * GDK);
        qf[0].q[0] = qp[hi];      qf[0].q[1] = qp[2 + hi];
        qf[1].q[0] = qp[4 + hi];  qf[1].q[1] = qp[6 + hi];
    }

    v8f o[4];
#pragma unroll
    for (int t = 0; t < 4; ++t) o[t] = (v8f){0.f,0.f,0.f,0.f,0.f,0.f,0.f,0.f};
    float mrow[8], lrow[8];
#pragma unroll
    for (int r = 0; r < 8; ++r) { mrow[r] = NEG_BIG; lrow[r] = 0.f; }

    const int srow = tid >> 1;          // stager: 64 rows x (2 x 32) cols
    const int scol = (tid & 1) * 32;
    const _Float16* gK = Kp + (size_t)srow * GDK + scol;  // + kv0*GDK
    const _Float16* gV = Vp + (size_t)srow * GS + scol;   // + kv0

    // prologue: stage chunk 0 into buffer 0
    async_copy64B(LDS_OFF(&Ks[0][srow][scol]), gK);
    async_copy64B(LDS_OFF(&Vs[0][srow][scol]), gV);
    ASYNC_WAIT0();
    __syncthreads();

    int buf = 0;
    const int nchunk = qt + 1;  // causal
    for (int c = 0; c < nchunk; ++c) {
        const int kv0 = c * 64;
        if (c + 1 < nchunk) {  // prefetch next chunk (overlaps WMMA + softmax)
            async_copy64B(LDS_OFF(&Ks[buf ^ 1][srow][scol]), gK + (size_t)(kv0 + 64) * GDK);
            async_copy64B(LDS_OFF(&Vs[buf ^ 1][srow][scol]), gV + kv0 + 64);
        }

        // S = Q @ K^T  (4 kv tiles x 2 dk K-steps)
        v8f st[4];
#pragma unroll
        for (int t = 0; t < 4; ++t) {
            v8f s = (v8f){0.f,0.f,0.f,0.f,0.f,0.f,0.f,0.f};
#pragma unroll
            for (int kk = 0; kk < 2; ++kk) {
                F16Frag bf;  // B: col = kv pos, contraction = dk
                bf.q[0] = *(const uint4*)&Ks[buf][t * 16 + l15][kk * 32 + hi * 8];
                bf.q[1] = *(const uint4*)&Ks[buf][t * 16 + l15][kk * 32 + 16 + hi * 8];
                s = WMMA16x16x32(qf[kk].v, bf.v, s);
            }
            st[t] = s;
        }
        // causal mask
#pragma unroll
        for (int t = 0; t < 4; ++t)
#pragma unroll
            for (int r = 0; r < 8; ++r) {
                const int row = q0 + hi * 8 + r, col = kv0 + t * 16 + l15;
                if (col > row) st[t][r] = NEG_BIG;
            }
        // online softmax (row lives across 16 lanes of one half, vgpr r)
#pragma unroll
        for (int r = 0; r < 8; ++r) {
            float mx = st[0][r];
#pragma unroll
            for (int t = 1; t < 4; ++t) mx = fmaxf(mx, st[t][r]);
#pragma unroll
            for (int off = 1; off < 16; off <<= 1) mx = fmaxf(mx, __shfl_xor(mx, off, 32));
            const float mn = fmaxf(mrow[r], mx);
            const float alpha = __expf(mrow[r] - mn);
            mrow[r] = mn;
            lrow[r] *= alpha;
#pragma unroll
            for (int t = 0; t < 4; ++t) o[t][r] *= alpha;
            float sum = 0.f;
#pragma unroll
            for (int t = 0; t < 4; ++t) {
                const float p = __expf(st[t][r] - mn);
                st[t][r] = p;
                sum += p;
            }
#pragma unroll
            for (int off = 1; off < 16; off <<= 1) sum += __shfl_xor(sum, off, 32);
            lrow[r] += sum;
        }
        // C-layout -> A-layout via per-wave LDS
#pragma unroll
        for (int t = 0; t < 4; ++t)
#pragma unroll
            for (int r = 0; r < 8; ++r)
                Ps[wave][hi * 8 + r][t * 16 + l15] = (_Float16)st[t][r];
        __syncthreads();
        // O += P @ V  (2 kv K-steps x 4 dk tiles)
#pragma unroll
        for (int kk = 0; kk < 2; ++kk) {
            F16Frag pf;
            pf.q[0] = *(const uint4*)&Ps[wave][l15][kk * 32 + hi * 8];
            pf.q[1] = *(const uint4*)&Ps[wave][l15][kk * 32 + 16 + hi * 8];
#pragma unroll
            for (int t = 0; t < 4; ++t) {
                F16Frag vf;  // B: col = dk, contraction = kv
                vf.q[0] = *(const uint4*)&Vs[buf][t * 16 + l15][kk * 32 + hi * 8];
                vf.q[1] = *(const uint4*)&Vs[buf][t * 16 + l15][kk * 32 + 16 + hi * 8];
                o[t] = WMMA16x16x32(pf.v, vf.v, o[t]);
            }
        }
        ASYNC_WAIT0();
        __syncthreads();
        buf ^= 1;
    }

    // write merged-head f16 output [b*S + row, h*64 + dk]
    const int b = bh >> 4, h = bh & 15;
#pragma unroll
    for (int t = 0; t < 4; ++t)
#pragma unroll
        for (int r = 0; r < 8; ++r) {
            const int row = q0 + hi * 8 + r;
            const float val = o[t][r] / lrow[r];
            Aout[(size_t)(b * GS + row) * GD + h * GDK + t * 16 + l15] = (_Float16)val;
        }
}

// ---------------------------------------------------------------------------
extern "C" void kernel_launch(void* const* d_in, const int* in_sizes, int n_in,
                              void* d_out, int out_size, void* d_ws, size_t ws_size,
                              hipStream_t stream) {
    const float* hs     = (const float*)d_in[0];  // [2,2048,1024]
    const float* W_attn = (const float*)d_in[1];  // [1024,3072]
    const float* b_attn = (const float*)d_in[2];  // [3072]
    const float* W_proj = (const float*)d_in[3];  // [1024,1024]
    const float* b_proj = (const float*)d_in[4];  // [1024]
    float* out = (float*)d_out;                   // [2,2048,1024]

    const size_t nHS = (size_t)GB * GS * GD;
    const size_t nWA = (size_t)GD * 3 * GD;
    const size_t nWP = (size_t)GD * GD;
    const size_t nQ  = (size_t)GB * GH * GS * GDK;

    _Float16* hsH   = (_Float16*)d_ws;
    _Float16* WaT   = hsH + nHS;   // [3072][1024] (W_attn^T)
    _Float16* WpT   = WaT + nWA;   // [1024][1024] (W_proj^T)
    _Float16* Qws   = WpT + nWP;   // [B,H,S,64]
    _Float16* Kws   = Qws + nQ;    // [B,H,S,64]
    _Float16* Vws   = Kws + nQ;    // [B,H,64,S]
    _Float16* AttnH = Vws + nQ;    // [B*S, D]

    cvt_f32_to_f16<<<(int)(nHS / 256), 256, 0, stream>>>(hs, hsH, (int)nHS);
    cvt_f32_to_f16_T<<<dim3(3 * GD / 32, GD / 32), 256, 0, stream>>>(W_attn, WaT, GD, 3 * GD);
    cvt_f32_to_f16_T<<<dim3(GD / 32, GD / 32), 256, 0, stream>>>(W_proj, WpT, GD, GD);

    gemm_f16_wmma<<<dim3(3 * GD / 128, GB * GS / 128), 256, 0, stream>>>(
        hsH, WaT, b_attn, nullptr, Qws, Kws, Vws, GB * GS, 3 * GD, GD, 1);

    flash_attn_wmma<<<dim3(GB * GH, GS / 64), 128, 0, stream>>>(Qws, Kws, Vws, AttnH);

    gemm_f16_wmma<<<dim3(GD / 128, GB * GS / 128), 256, 0, stream>>>(
        AttnH, WpT, b_proj, out, nullptr, nullptr, nullptr, GB * GS, GD, GD, 0);
}
